// SimpleCapsNet_88991722373501
// MI455X (gfx1250) — compile-verified, matrix operand order checked by reading, not run on hardware
//
#include <hip/hip_runtime.h>
#include <hip/hip_bf16.h>
#include <math.h>

typedef __attribute__((ext_vector_type(16))) _Float16 v16h;
typedef __attribute__((ext_vector_type(8)))  float    v8f;
typedef __attribute__((ext_vector_type(2)))  int      v2i_t;

#define EPSF 1e-7f

// ---------------------------------------------------------------------------
// gfx1250 async global->LDS copy (probe-confirmed signature:
//   (AS1 v2i* gsrc, AS3 v2i* ldsdst, imm offset, imm cpol))
// ---------------------------------------------------------------------------
#if defined(__has_builtin)
#  if __has_builtin(__builtin_amdgcn_global_load_async_to_lds_b64) && \
      __has_builtin(__builtin_amdgcn_s_wait_asynccnt)
#    define USE_ASYNC_LDS 1
#  endif
#endif
#ifndef USE_ASYNC_LDS
#  define USE_ASYNC_LDS 0
#endif

__device__ __forceinline__ void copy8(_Float16* l, const _Float16* g) {
#if USE_ASYNC_LDS
    __builtin_amdgcn_global_load_async_to_lds_b64(
        (__attribute__((address_space(1))) v2i_t*)g,
        (__attribute__((address_space(3))) v2i_t*)l, 0, 0);
#else
    *(unsigned long long*)l = *(const unsigned long long*)g;
#endif
}
__device__ __forceinline__ void wait_async_all() {      // drain everything
#if USE_ASYNC_LDS
    __builtin_amdgcn_s_wait_asynccnt(0);
#endif
}
__device__ __forceinline__ void wait_async_prev() {     // keep 6 in flight
#if USE_ASYNC_LDS
    __builtin_amdgcn_s_wait_asynccnt(6);
#endif
}

// ---------------------------------------------------------------------------
// A-tile mappers over f16 sources. Contract: the 32-element K-slice
// [stepoff(kbase), +32) of any row is CONTIGUOUS in memory.
// ---------------------------------------------------------------------------
struct F16MapConv {   // NHWC activation [B,H,W,Cin], weights packed k=(fy,fx,ci)
    const _Float16* src; int H, W, Cin, Cin_log2, KW, stride, OW, OHW;
    __device__ size_t rowbase(int m) const {      // once per block (hoisted divs)
        int bb = m / OHW; int rem = m - bb * OHW;
        int oy = rem / OW, ox = rem - oy * OW;
        return ((size_t)(bb * H + oy * stride) * W + ox * stride) * Cin;
    }
    __device__ int stepoff(int kbase) const {     // wave-uniform, per K-step
        int fyfx = kbase >> Cin_log2;             // Cin multiple of 32 => exact
        int fy = fyfx / KW, fx = fyfx - fy * KW;
        return (fy * W + fx) * Cin + (kbase & (Cin - 1));
    }
};
struct F16MapRow {    // plain row-major f16 [M,K]
    const _Float16* src; int K;
    __device__ size_t rowbase(int m) const { return (size_t)m * K; }
    __device__ int stepoff(int kbase) const { return kbase; }
};

// ---------------------------------------------------------------------------
// WMMA GEMM: C[M,N] = act(A[M,K]*B[K,N] + bias), A/B f16, accum f32.
// Block 256 thr = 8 waves; block tile 128x64; wave tile 32x32 (4 WMMA/step).
// Double-buffered LDS: async copies for step k+1 fly behind step k's WMMAs;
// s_wait_asynccnt 6 drains only the previous tile's 6 copies per wave.
// B(k,n) = Bw[n*Kpad + k] (pre-packed f16). ACT: 1=relu 2=sigmoid
// ---------------------------------------------------------------------------
template <int ACT, typename AMap>
__global__ __launch_bounds__(256)
void gemm_f16(AMap amap, const _Float16* __restrict__ Bw,
              const float* __restrict__ bias,
              float* __restrict__ out32, _Float16* __restrict__ out16,
              int N, int Kpad, int Ksteps) {
    __shared__ _Float16 As[2][128][40];  // 80B row stride: 8B aligned
    __shared__ _Float16 Bs[2][64][40];   // stored [n][k]

    const int t = threadIdx.x, lane = t & 31, wave = t >> 5;
    const int m0 = blockIdx.x * 128, n0 = blockIdx.y * 64;
    const int wm = (wave & 3) * 32;      // wave M sub-tile
    const int wn = (wave >> 2) * 32;     // wave N sub-tile
    const int half8 = (lane & 16) >> 1;
    const int lrow  = lane & 15;

    // staging: thread t copies 8B chunks; row = t>>3 (+32*i), halves (t&7)*4..+3
    const int srow = t >> 3;
    const int skc  = (t & 7) * 4;
    size_t abase[4];
#pragma unroll
    for (int i = 0; i < 4; ++i) abase[i] = amap.rowbase(m0 + srow + 32 * i);
    size_t bbase[2];
#pragma unroll
    for (int i = 0; i < 2; ++i) bbase[i] = (size_t)(n0 + srow + 32 * i) * Kpad;

    // 6 async copy instructions per wave per stage (4 A + 2 B)
    auto stage = [&](int buf, int kbase) {
        const int aoff = amap.stepoff(kbase) + skc;
#pragma unroll
        for (int i = 0; i < 4; ++i)
            copy8(&As[buf][srow + 32 * i][skc], amap.src + abase[i] + aoff);
#pragma unroll
        for (int i = 0; i < 2; ++i)
            copy8(&Bs[buf][srow + 32 * i][skc], Bw + bbase[i] + kbase + skc);
    };

    v8f a00 = {0,0,0,0,0,0,0,0}, a01 = a00, a10 = a00, a11 = a00;

    stage(0, 0);                                    // prologue fill
    for (int ks = 0; ks < Ksteps; ++ks) {
        const int cur = ks & 1;
        if (ks + 1 < Ksteps) {
            stage(cur ^ 1, (ks + 1) * 32);          // issue next tile (async)
            wait_async_prev();                      // drain only current tile
        } else {
            wait_async_all();
        }
        if (ks + 2 < Ksteps) {                      // prefetch two steps ahead
            const int aoffn = amap.stepoff((ks + 2) * 32) + skc;
#pragma unroll
            for (int i = 0; i < 4; ++i)
                __builtin_prefetch(amap.src + abase[i] + aoffn, 0, 3);
        }
        __syncthreads();                            // all waves' copies landed

        // fragment K-mapping per ISA 16-bit A 16x32 layout (B mirrored):
        // lanes 0-15: K 0-7 (e<8), 16-23 (e>=8); lanes 16-31: +8
        v16h f_a0, f_a1, f_b0, f_b1;
#pragma unroll
        for (int e = 0; e < 16; ++e) {
            int k = ((e & 8) ? (e + 8) : e) + half8;
            f_a0[e] = As[cur][wm + lrow][k];
            f_a1[e] = As[cur][wm + 16 + lrow][k];
            f_b0[e] = Bs[cur][wn + lrow][k];
            f_b1[e] = Bs[cur][wn + 16 + lrow][k];
        }
        a00 = __builtin_amdgcn_wmma_f32_16x16x32_f16(false, f_a0, false, f_b0, (short)0, a00, false, false);
        a01 = __builtin_amdgcn_wmma_f32_16x16x32_f16(false, f_a0, false, f_b1, (short)0, a01, false, false);
        a10 = __builtin_amdgcn_wmma_f32_16x16x32_f16(false, f_a1, false, f_b0, (short)0, a10, false, false);
        a11 = __builtin_amdgcn_wmma_f32_16x16x32_f16(false, f_a1, false, f_b1, (short)0, a11, false, false);
        __syncthreads();                            // protect buffer reuse at ks+2
    }

    // epilogue: C/D layout = VGPR r -> row r (lanes 0-15) / row r+8 (lanes 16-31)
    auto store_tile = [&](const v8f& acc, int mi, int ni) {
        const int col = n0 + wn + ni + lrow;
        const float bv = bias[col];
#pragma unroll
        for (int r = 0; r < 8; ++r) {
            const int row = m0 + wm + mi + r + half8;
            float v = acc[r] + bv;
            if (ACT == 1) v = fmaxf(v, 0.f);
            else if (ACT == 2) v = 1.f / (1.f + __expf(-v));
            const size_t o = (size_t)row * N + col;
            if (out32) out32[o] = v;
            if (out16) out16[o] = (_Float16)v;
        }
    };
    store_tile(a00, 0, 0);  store_tile(a01, 0, 16);
    store_tile(a10, 16, 0); store_tile(a11, 16, 16);
}

// ---------------------------------------------------------------------------
// One-time packing kernels (per launch; weights are runtime inputs)
// ---------------------------------------------------------------------------
__global__ void pack_rowmajor_f16(const float* __restrict__ w, _Float16* __restrict__ dst,
                                  int K, int Kpad, int total /* N*Kpad */) {
    int idx = blockIdx.x * 256 + threadIdx.x;
    if (idx >= total) return;
    int n = idx / Kpad, k = idx - n * Kpad;
    dst[idx] = (_Float16)((k < K) ? w[(size_t)n * K + k] : 0.f);
}
__global__ void pack_conv_f16(const float* __restrict__ w, _Float16* __restrict__ dst,
                              int Cin, int Cin_log2, int KHW, int total /* N*K */) {
    // OIHW -> [n][ (fy*KW+fx)*Cin + ci ]
    int idx = blockIdx.x * 256 + threadIdx.x;
    if (idx >= total) return;
    int K = Cin * KHW;
    int n = idx / K, k = idx - n * K;
    int fyfx = k >> Cin_log2, ci = k & (Cin - 1);
    dst[idx] = (_Float16)w[((size_t)n * Cin + ci) * KHW + fyfx];
}
__global__ void im2col_conv1(const float* __restrict__ x, _Float16* __restrict__ A1) {
    // A1: [1024*225][32] f16, k=(ci,fy,fx) matching OIHW row-major; k>=27 -> 0
    size_t idx = (size_t)blockIdx.x * 256 + threadIdx.x;      // over 230400*32
    if (idx >= (size_t)230400 * 32) return;
    int k = (int)(idx & 31);
    size_t m = idx >> 5;
    float v = 0.f;
    if (k < 27) {
        int bb = (int)(m / 225); int rem = (int)(m - (size_t)bb * 225);
        int oy = rem / 15, ox = rem - oy * 15;
        int ci = k / 9, r = k - ci * 9, fy = r / 3, fx = r - fy * 3;
        v = x[(((size_t)bb * 3 + ci) * 32 + oy * 2 + fy) * 32 + ox * 2 + fx];
    }
    A1[idx] = (_Float16)v;
}

// ---------------------------------------------------------------------------
// Per-batch primary-caps squash + u_hat + 3-iter dynamic routing (VALU; tiny)
// ---------------------------------------------------------------------------
__global__ __launch_bounds__(256)
void routing_kernel(const float* __restrict__ c3,      // [B,2,2,256] NHWC f32
                    const float* __restrict__ caps_w,  // [64,10,16,16]
                    const float* __restrict__ caps_b,  // [10,16]
                    const int* __restrict__ y,
                    float* __restrict__ v_len,         // [B,10] (d_out)
                    _Float16* __restrict__ masked) {   // [B,160] f16 -> decoder
    __shared__ float prim[64][16];
    __shared__ float uhat[64][10][16];
    __shared__ float bij[64][10];
    __shared__ float cij[64][10];
    __shared__ float sv[10][16];
    __shared__ float vv[10][16];

    const int b = blockIdx.x, t = threadIdx.x;

    for (int f = t; f < 1024; f += 256) {   // NCHW flat view from NHWC storage
        int c = f >> 2, sp = f & 3, yy = sp >> 1, xx = sp & 1;
        prim[f >> 4][f & 15] = c3[(((size_t)b * 2 + yy) * 2 + xx) * 256 + c];
    }
    __syncthreads();
    if (t < 64) {
        float dot = 0.f;
        for (int k = 0; k < 16; ++k) dot += prim[t][k] * prim[t][k];
        float sc = dot / (1.f + dot) * rsqrtf(dot + EPSF);
        for (int k = 0; k < 16; ++k) prim[t][k] *= sc;
    }
    __syncthreads();
    for (int idx = t; idx < 64 * 10 * 16; idx += 256) {
        int i = idx / 160, o = (idx / 16) % 10, j = idx & 15;
        const float* w = caps_w + (((size_t)i * 10 + o) * 16) * 16 + j;
        float s = 0.f;
        for (int k = 0; k < 16; ++k) s += prim[i][k] * w[k * 16];
        uhat[i][o][j] = s;
    }
    for (int idx = t; idx < 640; idx += 256) bij[idx / 10][idx % 10] = 0.f;
    __syncthreads();

    for (int r = 0; r < 3; ++r) {
        if (t < 64) {
            float mx = -1e30f;
            for (int o = 0; o < 10; ++o) mx = fmaxf(mx, bij[t][o]);
            float e[10], sum = 0.f;
            for (int o = 0; o < 10; ++o) { e[o] = __expf(bij[t][o] - mx); sum += e[o]; }
            float inv = 1.f / sum;
            for (int o = 0; o < 10; ++o) cij[t][o] = e[o] * inv;
        }
        __syncthreads();
        if (t < 160) {
            int o = t >> 4, j = t & 15;
            float s = caps_b[o * 16 + j];
            for (int i = 0; i < 64; ++i) s += cij[i][o] * uhat[i][o][j];
            sv[o][j] = s;
        }
        __syncthreads();
        if (t < 10) {
            float dot = 0.f;
            for (int j = 0; j < 16; ++j) dot += sv[t][j] * sv[t][j];
            float sc = dot / (1.f + dot) * rsqrtf(dot + EPSF);
            for (int j = 0; j < 16; ++j) vv[t][j] = sv[t][j] * sc;
        }
        __syncthreads();
        if (r < 2) {
            for (int idx = t; idx < 640; idx += 256) {
                int i = idx / 10, o = idx % 10;
                float ag = 0.f;
                for (int j = 0; j < 16; ++j) ag += uhat[i][o][j] * vv[o][j];
                bij[i][o] += ag;
            }
            __syncthreads();
        }
    }

    if (t < 10) {
        float dot = 0.f;
        for (int j = 0; j < 16; ++j) dot += vv[t][j] * vv[t][j];
        v_len[(size_t)b * 10 + t] = sqrtf(dot + EPSF);
    }
    int lbl = y[b];
    for (int idx = t; idx < 160; idx += 256) {
        int o = idx >> 4;
        masked[(size_t)b * 160 + idx] = (_Float16)((o == lbl) ? vv[o][idx & 15] : 0.f);
    }
}

// ---------------------------------------------------------------------------
extern "C" void kernel_launch(void* const* d_in, const int* in_sizes, int n_in,
                              void* d_out, int out_size, void* d_ws, size_t ws_size,
                              hipStream_t stream) {
    const float* x       = (const float*)d_in[0];
    const int*   y       = (const int*)  d_in[1];
    const float* conv1_w = (const float*)d_in[2];
    const float* conv1_b = (const float*)d_in[3];
    const float* conv2_w = (const float*)d_in[4];
    const float* conv2_b = (const float*)d_in[5];
    const float* conv3_w = (const float*)d_in[6];
    const float* conv3_b = (const float*)d_in[7];
    const float* caps_w  = (const float*)d_in[8];
    const float* caps_b  = (const float*)d_in[9];
    const float* fc1_w   = (const float*)d_in[10];
    const float* fc1_b   = (const float*)d_in[11];
    const float* fc2_w   = (const float*)d_in[12];
    const float* fc2_b   = (const float*)d_in[13];
    const float* fc3_w   = (const float*)d_in[14];
    const float* fc3_b   = (const float*)d_in[15];

    float* vlen = (float*)d_out;              // [1024,10]
    float* dec  = (float*)d_out + 1024 * 10;  // [1024,3072]

    // bump allocator over d_ws (256B aligned)
    char* wsp = (char*)d_ws;
    auto alloc = [&](size_t bytes) -> char* {
        char* p = wsp; wsp += (bytes + 255) & ~(size_t)255; return p;
    };
    _Float16* A1     = (_Float16*)alloc((size_t)230400 * 32 * 2);
    _Float16* c1     = (_Float16*)alloc((size_t)1024 * 225 * 128 * 2);
    _Float16* c2     = (_Float16*)alloc((size_t)1024 * 36 * 256 * 2);
    float*    c3     = (float*)   alloc((size_t)1024 * 4 * 256 * 4);
    _Float16* masked = (_Float16*)alloc((size_t)1024 * 160 * 2);
    _Float16* h1     = (_Float16*)alloc((size_t)1024 * 512 * 2);
    _Float16* h2     = (_Float16*)alloc((size_t)1024 * 1024 * 2);
    _Float16* w1p    = (_Float16*)alloc((size_t)128 * 32 * 2);
    _Float16* w2p    = (_Float16*)alloc((size_t)256 * 3200 * 2);
    _Float16* w3p    = (_Float16*)alloc((size_t)256 * 6400 * 2);
    _Float16* w4p    = (_Float16*)alloc((size_t)512 * 160 * 2);
    _Float16* w5p    = (_Float16*)alloc((size_t)1024 * 512 * 2);
    _Float16* w6p    = (_Float16*)alloc((size_t)3072 * 1024 * 2);

    dim3 blk(256);
    auto cdiv = [](int a, int b) { return (a + b - 1) / b; };

    // ---- pack weights + conv1 im2col ----
    pack_rowmajor_f16<<<cdiv(128 * 32, 256), blk, 0, stream>>>(conv1_w, w1p, 27, 32, 128 * 32);
    pack_conv_f16<<<cdiv(256 * 3200, 256), blk, 0, stream>>>(conv2_w, w2p, 128, 7, 25, 256 * 3200);
    pack_conv_f16<<<cdiv(256 * 6400, 256), blk, 0, stream>>>(conv3_w, w3p, 256, 8, 25, 256 * 6400);
    pack_rowmajor_f16<<<cdiv(512 * 160, 256), blk, 0, stream>>>(fc1_w, w4p, 160, 160, 512 * 160);
    pack_rowmajor_f16<<<cdiv(1024 * 512, 256), blk, 0, stream>>>(fc2_w, w5p, 512, 512, 1024 * 512);
    pack_rowmajor_f16<<<cdiv(3072 * 1024, 256), blk, 0, stream>>>(fc3_w, w6p, 1024, 1024, 3072 * 1024);
    im2col_conv1<<<cdiv(230400 * 32, 256), blk, 0, stream>>>(x, A1);

    // ---- conv1: M=230400 N=128 K=32 ----
    {
        F16MapRow m{A1, 32};
        gemm_f16<1, F16MapRow><<<dim3(230400 / 128, 128 / 64), blk, 0, stream>>>(
            m, w1p, conv1_b, nullptr, c1, 128, 32, 1);
    }
    // ---- conv2: M=36864 N=256 K=3200 ----
    {
        F16MapConv m{c1, 15, 15, 128, 7, 5, 2, 6, 36};
        gemm_f16<1, F16MapConv><<<dim3(36864 / 128, 256 / 64), blk, 0, stream>>>(
            m, w2p, conv2_b, nullptr, c2, 256, 3200, 100);
    }
    // ---- conv3: M=4096 N=256 K=6400 (f32 out for routing) ----
    {
        F16MapConv m{c2, 6, 6, 256, 8, 5, 1, 2, 4};
        gemm_f16<1, F16MapConv><<<dim3(4096 / 128, 256 / 64), blk, 0, stream>>>(
            m, w3p, conv3_b, c3, nullptr, 256, 6400, 200);
    }
    // ---- squash + u_hat + routing + mask ----
    routing_kernel<<<dim3(1024), blk, 0, stream>>>(c3, caps_w, caps_b, y, vlen, masked);

    // ---- decoder ----
    {
        F16MapRow m{masked, 160};
        gemm_f16<1, F16MapRow><<<dim3(1024 / 128, 512 / 64), blk, 0, stream>>>(
            m, w4p, fc1_b, nullptr, h1, 512, 160, 5);
    }
    {
        F16MapRow m{h1, 512};
        gemm_f16<1, F16MapRow><<<dim3(1024 / 128, 1024 / 64), blk, 0, stream>>>(
            m, w5p, fc2_b, nullptr, h2, 1024, 512, 16);
    }
    {
        F16MapRow m{h2, 1024};
        gemm_f16<2, F16MapRow><<<dim3(1024 / 128, 3072 / 64), blk, 0, stream>>>(
            m, w6p, fc3_b, dec, nullptr, 3072, 1024, 32);
    }
}
